// ISTFT_10599979287129
// MI455X (gfx1250) — matrix operand order repaired and checked by reading, not compiled, not run on hardware
//
#include <hip/hip_runtime.h>

// CDNA5 / gfx1250 ISTFT: fused (stft^T @ basis) + overlap-add + window
// normalization as a single bf16x3 WMMA GEMM (M=16x2047, N=256, K=4104).
// Double-buffered LDS + software pipeline. Basis pre-packed to bf16 hi/lo
// tiles in workspace; B tiles then stream via GLOBAL_LOAD_ASYNC_TO_LDS_B128
// (ASYNCcnt) with zero VGPR staging.

typedef __attribute__((ext_vector_type(16))) __bf16 v16bf;
typedef __attribute__((ext_vector_type(8)))  float  v8f;

#define FILTER_LEN 1024
#define NCH        1026          // filter_len + 2
#define FRAMES     2048
#define BATCH      16
#define KTOT       (NCH * 4)     // 4104 (kappa = 4*c + j)
#define KSTEPS     129           // ceil(4104 / 32)
#define OUT_PER_B  524032        // (FRAMES-1)*256 + 1024 - 2*512
#define MT         64            // h-rows per block
#define ASTRIDE    40            // LDS row stride in bf16 (bank spread, 16B-aligned)
#define BSTRIDE    40
#define BPLANE     (256 * BSTRIDE)       // one hi or lo plane, bf16 elems
#define BTILE      (2 * BPLANE)          // hi+lo planes per K-step (40960 B)

// global_load_async_to_lds_b128: VDST = LDS byte address (VGPR),
// VADDR = 64-bit global address, SADDR = off.  INST_OFFSET adds to BOTH
// the LDS and global addresses (ISA 08_async_tensor §4.4), so one base
// covers the whole per-thread chunk set.
#define ASYNC_B128(ldsv, gptr, OFF)                                          \
  asm volatile("global_load_async_to_lds_b128 %0, %1, off offset:" #OFF      \
               :: "v"(ldsv), "v"(gptr) : "memory")

__device__ __forceinline__ void wait_asynccnt0() {
  asm volatile("s_wait_asynccnt 0x0" ::: "memory");
}

__global__ void pack_basis_kernel(const float* __restrict__ basis,
                                  __bf16* __restrict__ Bpack) {
  int idx = blockIdx.x * 256 + threadIdx.x;
  const int total = KSTEPS * 256 * BSTRIDE;
  if (idx >= total) return;
  int kl   = idx % BSTRIDE;
  int n    = (idx / BSTRIDE) % 256;
  int step = idx / (BSTRIDE * 256);
  int kk   = step * 32 + kl;
  float x  = (kl < 32 && kk < KTOT) ? basis[kk * 256 + n] : 0.0f;
  __bf16 hi = (__bf16)x;
  __bf16 lo = (__bf16)(x - (float)hi);
  size_t base = (size_t)step * BTILE;
  Bpack[base + n * BSTRIDE + kl]          = hi;
  Bpack[base + BPLANE + n * BSTRIDE + kl] = lo;
}

template <bool USEWS>
__global__ __launch_bounds__(256)
void istft_wmma_kernel(const float* __restrict__ stft,
                       const float* __restrict__ basis,   // viewed as [4104][256]
                       const __bf16* __restrict__ Bpack,  // [129][2][256][40] or null
                       const float* __restrict__ sqwin,
                       float* __restrict__ out) {
  __shared__ __align__(16) __bf16 Asm[2][2][MT * ASTRIDE];  // [buf][hi/lo]
  __shared__ __align__(16) __bf16 Bsm[2][BTILE];            // [buf][hi|lo planes]
  __shared__ float swl[FILTER_LEN];

  const int tid = threadIdx.x;
  const int b   = blockIdx.x;
  const int h0  = 2 + (int)blockIdx.y * MT;
  const float* stftB = stft + (size_t)b * NCH * FRAMES;

  for (int i = tid; i < FILTER_LEN; i += 256) swl[i] = sqwin[i];

  const int lane  = tid & 31;
  const int w     = tid >> 5;       // 8 waves
  const int msub  = w >> 1;         // 4 M-subtiles of 16
  const int nhalf = w & 1;          // 2 N-halves of 128
  const int nloc  = lane & 15;
  const int g16   = lane >> 4;

  // A staging: thread owns row m = tid&63, 8 contiguous kappa at group tid>>6
  const int am  = tid & 63;
  const int akg = tid >> 6;

  v8f acc[8] = {};

  float aReg[8];
  float bReg[32];   // !USEWS only: fp32 basis column

  auto loadA = [&](int step) {
    #pragma unroll
    for (int e = 0; e < 8; ++e) {
      int kl = akg * 8 + e;
      int c  = step * 8 + (kl >> 2);
      int f  = h0 + am - (kl & 3);
      aReg[e] = (c < NCH && f >= 0 && f < FRAMES) ? stftB[c * FRAMES + f] : 0.0f;
    }
  };
  // USEWS: fire-and-forget async copy of one pre-packed 40 KB B tile to LDS.
  auto asyncB = [&](int step, int buf) {
    unsigned ldsv = (unsigned)(size_t)(&Bsm[buf][0]) + (unsigned)tid * 16u;
    const char* g = (const char*)(Bpack + (size_t)step * BTILE) + (size_t)tid * 16u;
    ASYNC_B128(ldsv, g, 0);
    ASYNC_B128(ldsv, g, 4096);
    ASYNC_B128(ldsv, g, 8192);
    ASYNC_B128(ldsv, g, 12288);
    ASYNC_B128(ldsv, g, 16384);
    ASYNC_B128(ldsv, g, 20480);
    ASYNC_B128(ldsv, g, 24576);
    ASYNC_B128(ldsv, g, 28672);
    ASYNC_B128(ldsv, g, 32768);
    ASYNC_B128(ldsv, g, 36864);
  };
  auto loadB = [&](int step) {   // !USEWS
    #pragma unroll
    for (int kl = 0; kl < 32; ++kl) {
      int kk = step * 32 + kl;
      bReg[kl] = (kk < KTOT) ? basis[kk * 256 + tid] : 0.0f;
    }
  };
  auto storeA = [&](int cur) {
    union { __bf16 h[8]; uint4 q; } uh, ul;
    #pragma unroll
    for (int e = 0; e < 8; ++e) {
      __bf16 hi = (__bf16)aReg[e];
      uh.h[e] = hi;
      ul.h[e] = (__bf16)(aReg[e] - (float)hi);
    }
    *(uint4*)&Asm[cur][0][am * ASTRIDE + akg * 8] = uh.q;
    *(uint4*)&Asm[cur][1][am * ASTRIDE + akg * 8] = ul.q;
  };
  auto storeB = [&](int cur) {   // !USEWS
    #pragma unroll
    for (int g = 0; g < 4; ++g) {
      union { __bf16 h[8]; uint4 q; } uh, ul;
      #pragma unroll
      for (int e = 0; e < 8; ++e) {
        float x = bReg[g * 8 + e];
        __bf16 hi = (__bf16)x;
        uh.h[e] = hi;
        ul.h[e] = (__bf16)(x - (float)hi);
      }
      *(uint4*)&Bsm[cur][tid * BSTRIDE + g * 8]          = uh.q;
      *(uint4*)&Bsm[cur][BPLANE + tid * BSTRIDE + g * 8] = ul.q;
    }
  };
  auto compute = [&](int cur) {
    const int aoff = (msub * 16 + nloc) * ASTRIDE + g16 * 8;
    v16bf ah, al;
    {
      uint4* p = (uint4*)&ah;
      p[0] = *(const uint4*)&Asm[cur][0][aoff];
      p[1] = *(const uint4*)&Asm[cur][0][aoff + 16];
      uint4* q = (uint4*)&al;
      q[0] = *(const uint4*)&Asm[cur][1][aoff];
      q[1] = *(const uint4*)&Asm[cur][1][aoff + 16];
    }
    #pragma unroll
    for (int nt = 0; nt < 8; ++nt) {
      int bo = (nhalf * 128 + nt * 16 + nloc) * BSTRIDE + g16 * 8;
      v16bf bh, bl;
      uint4* p = (uint4*)&bh;
      p[0] = *(const uint4*)&Bsm[cur][bo];
      p[1] = *(const uint4*)&Bsm[cur][bo + 16];
      uint4* q = (uint4*)&bl;
      q[0] = *(const uint4*)&Bsm[cur][BPLANE + bo];
      q[1] = *(const uint4*)&Bsm[cur][BPLANE + bo + 16];
      // bf16x3: a_hi*b_hi + a_hi*b_lo + a_lo*b_hi  (~fp32 accuracy)
      acc[nt] = __builtin_amdgcn_wmma_f32_16x16x32_bf16(false, ah, false, bh,
                                                        (short)0, acc[nt], false, false);
      acc[nt] = __builtin_amdgcn_wmma_f32_16x16x32_bf16(false, ah, false, bl,
                                                        (short)0, acc[nt], false, false);
      acc[nt] = __builtin_amdgcn_wmma_f32_16x16x32_bf16(false, al, false, bh,
                                                        (short)0, acc[nt], false, false);
    }
  };

  // -------- software-pipelined main loop (double-buffered LDS) --------
  loadA(0);
  if constexpr (USEWS) asyncB(0, 0); else loadB(0);

  for (int step = 0; step < KSTEPS; ++step) {
    const int cur = step & 1;
    storeA(cur);
    if constexpr (USEWS) {
      wait_asynccnt0();                     // our B tile for buf[cur] landed
    } else {
      storeB(cur);
    }
    __syncthreads();                        // all waves staged buf[cur]; buf[1-cur] free
    if (step + 1 < KSTEPS) {                // next K-step: A -> regs, B -> async LDS,
      loadA(step + 1);                      // both overlap this step's WMMAs
      if constexpr (USEWS) asyncB(step + 1, 1 - cur); else loadB(step + 1);
    }
    compute(cur);
  }

  // -------- epilogue: window-envelope normalize, *4, crop, store --------
  float* outB = out + (size_t)b * OUT_PER_B;
  #pragma unroll
  for (int nt = 0; nt < 8; ++nt) {
    int r = nhalf * 128 + nt * 16 + nloc;
    #pragma unroll
    for (int v = 0; v < 8; ++v) {
      int m = v + (g16 << 3);               // C layout: VGPR v -> M = v + 8*(lane>=16)
      int h = h0 + msub * 16 + m;
      if (h > 2048) continue;
      float env = 1e-10f;
      #pragma unroll
      for (int j = 0; j < 4; ++j) {         // same validity mask as the A gather
        int f = h - j;
        if (f >= 0 && f < FRAMES) env += swl[j * 256 + r];
      }
      outB[(size_t)(h - 2) * 256 + r] = acc[nt][v] * 4.0f / env;
    }
  }
}

extern "C" void kernel_launch(void* const* d_in, const int* in_sizes, int n_in,
                              void* d_out, int out_size, void* d_ws, size_t ws_size,
                              hipStream_t stream) {
  (void)in_sizes; (void)n_in; (void)out_size;
  const float* stft  = (const float*)d_in[0];   // (16, 1026, 2048) f32
  const float* basis = (const float*)d_in[1];   // (1026, 1024) f32 == (4104, 256)
  const float* sqwin = (const float*)d_in[2];   // (1024,) f32
  float* out = (float*)d_out;                   // (16, 524032) f32

  dim3 grid(BATCH, 32);                         // 32 tiles of 64 hop-blocks: h in [2,2049]
  const size_t packBytes = (size_t)KSTEPS * BTILE * sizeof(__bf16);  // ~5.3 MB

  if (d_ws != nullptr && ws_size >= packBytes) {
    const int total  = KSTEPS * 256 * BSTRIDE;
    const int blocks = (total + 255) / 256;
    pack_basis_kernel<<<blocks, 256, 0, stream>>>(basis, (__bf16*)d_ws);
    istft_wmma_kernel<true><<<grid, 256, 0, stream>>>(
        stft, basis, (const __bf16*)d_ws, sqwin, out);
  } else {
    istft_wmma_kernel<false><<<grid, 256, 0, stream>>>(
        stft, basis, nullptr, sqwin, out);
  }
}
